// Architecture_50629074485965
// MI455X (gfx1250) — matrix operand order repaired and checked by reading, not compiled, not run on hardware
//
#include <hip/hip_runtime.h>
#include <hip/hip_bf16.h>
#include <math.h>

// ---------------------------------------------------------------------------
// Problem constants (fixed by the reference)
// ---------------------------------------------------------------------------
constexpr int B_   = 16;
constexpr int S_   = 512;
constexpr int D_   = 1024;
constexpr int H_   = 8;
constexpr int DFF_ = 4096;
constexpr int L_   = 3;
constexpr int DK_  = 128;       // D / H
constexpr int MM_  = B_ * S_;   // 8192 rows
constexpr float NEGV = -1e32f;

// ---------------------------------------------------------------------------
// Optional CDNA5 async global->LDS path (ASYNCcnt tracked)
// Builtin signature (from hipcc diagnostic): param0 = int4 __device__* (AS1,
// non-const, vector_size(16) pointee); 4 args total.
// ---------------------------------------------------------------------------
#if defined(__has_builtin)
#  if __has_builtin(__builtin_amdgcn_global_load_async_to_lds_b128) && \
      __has_builtin(__builtin_amdgcn_s_wait_asynccnt)
#    define USE_ASYNC_LDS 1
#  endif
#endif
#ifndef USE_ASYNC_LDS
#  define USE_ASYNC_LDS 0
#endif

#if USE_ASYNC_LDS
typedef int v4i_gcc __attribute__((vector_size(16)));
#define ASYNC_GPTR(p) ((v4i_gcc __attribute__((address_space(1)))*)(p))
#define ASYNC_LPTR(p) ((v4i_gcc __attribute__((address_space(3)))*)(p))
#endif

// ---------------------------------------------------------------------------
// WMMA helpers (CDNA5 / gfx1250, wave32): D = A(16x32 bf16) * B(32x16 bf16) + C(f32)
// ---------------------------------------------------------------------------
typedef __attribute__((ext_vector_type(16))) __bf16 v16bf;
typedef __attribute__((ext_vector_type(8)))  float  v8f;

union Frag {
    v16bf  v;
    __bf16 h[16];
    unsigned u[8];
};

union Pack2 {
    __bf16 h[2];
    unsigned u;
};

__device__ __forceinline__ v8f wmma_bf16(v16bf a, v16bf b, v8f c) {
    return __builtin_amdgcn_wmma_f32_16x16x32_bf16(false, a, false, b, (short)0, c,
                                                   false, false);
}

// A-operand fragment (16x32 row-major). lane%16 selects the row; khalf = lane/16.
// Per ISA layout: first-half lanes carry K {0..7, 16..23}, second half {8..15, 24..31}.
__device__ __forceinline__ v16bf load_a_frag_f32(const float* __restrict__ row, int khalf) {
    Frag f;
    const float* p0 = row + khalf * 8;
    const float* p1 = row + 16 + khalf * 8;
#pragma unroll
    for (int i = 0; i < 8; ++i) {
        f.h[i]     = (__bf16)p0[i];
        f.h[8 + i] = (__bf16)p1[i];
    }
    return f.v;
}

// B-operand fragment (32x16). lane%16 selects the column; loader takes a pointer
// to that column's 32 K-values contiguously. Lanes 0-15 carry K {0..7, 16..23}, 16-31 K {8..15, 24..31}.
__device__ __forceinline__ v16bf load_b_frag_f32(const float* __restrict__ col, int khalf) {
    Frag f;
    const float* p0 = col + khalf * 8;
    const float* p1 = col + 16 + khalf * 8;
#pragma unroll
    for (int i = 0; i < 8; ++i) {
        f.h[i]     = (__bf16)p0[i];
        f.h[8 + i] = (__bf16)p1[i];
    }
    return f.v;
}

__device__ __forceinline__ v16bf load_a_frag_lds(const __bf16* __restrict__ row, int khalf) {
    Frag f;
#pragma unroll
    for (int i = 0; i < 8; ++i) {
        f.h[i]     = row[khalf * 8 + i];
        f.h[8 + i] = row[16 + khalf * 8 + i];
    }
    return f.v;
}

__device__ __forceinline__ v16bf load_b_frag_lds(const __bf16* __restrict__ row, int khalf) {
    Frag f;
#pragma unroll
    for (int i = 0; i < 8; ++i) {
        f.h[i]     = row[khalf * 8 + i];
        f.h[8 + i] = row[16 + khalf * 8 + i];
    }
    return f.v;
}

// ---------------------------------------------------------------------------
// DAM Toeplitz bits: t[l,h,s] = 1 iff argmax over softmax(alphas - log(E+1e-5)) == 0
// ---------------------------------------------------------------------------
__global__ void dam_kernel(const float* __restrict__ alphas,
                           const float* __restrict__ E,
                           float* __restrict__ t, int n) {
    int i = blockIdx.x * blockDim.x + threadIdx.x;
    if (i >= n) return;
    float a0 = alphas[2 * i]     - logf(E[2 * i]     + 1e-5f);
    float a1 = alphas[2 * i + 1] - logf(E[2 * i + 1] + 1e-5f);
    t[i] = (a0 >= a1) ? 1.0f : 0.0f;
}

// ---------------------------------------------------------------------------
// GEMM: C[M,N] = act(A[M,K] @ W[N,K]^T + bias[N]); 128x128x32 tile, bf16 WMMA.
// 8 waves arranged 4(M) x 2(N): each wave owns a 32x64 sub-tile, B fragments
// reused across two A fragments. SPLIT==1 stores head-split (B,H,S,DK).
// ---------------------------------------------------------------------------
constexpr int BM = 128, BN = 128, BK = 32;
constexpr int LDA = 36; // bf16 stride pad -> conflict-free fragment reads

template <int ACT, int SPLIT>
__global__ __launch_bounds__(256) void gemm_kernel(const float* __restrict__ A,
                                                   const float* __restrict__ W,
                                                   const float* __restrict__ bias,
                                                   float* __restrict__ C,
                                                   int M, int N, int K) {
    __shared__ __bf16 As[BM * LDA];
    __shared__ __bf16 Ws[BN * LDA];
    const int tid = threadIdx.x;
    const int wid = tid >> 5, lane = tid & 31;
    const int khalf = lane >> 4, l16 = lane & 15;
    const int wrow = wid & 3;   // 4 M sub-slabs of 32 rows
    const int wcol = wid >> 2;  // 2 N halves of 64 cols
    const int m0 = blockIdx.y * BM, n0 = blockIdx.x * BN;

    v8f acc[2][4] = {};

    for (int k0 = 0; k0 < K; k0 += BK) {
        if (k0 + BK < K) { // prefetch next tiles (global_prefetch_b8)
            __builtin_prefetch(&A[(size_t)(m0 + (tid >> 1)) * K + k0 + BK], 0, 1);
            __builtin_prefetch(&W[(size_t)(n0 + (tid >> 1)) * K + k0 + BK], 0, 1);
        }
        // stage 128x32 f32 -> bf16 (float2 loads, packed dword LDS stores)
#pragma unroll
        for (int i = 0; i < 8; ++i) {
            int idx = tid + i * 256;        // 2048 pairs
            int r = idx >> 4, c = (idx & 15) * 2;
            float2 av = *(const float2*)&A[(size_t)(m0 + r) * K + k0 + c];
            float2 wv = *(const float2*)&W[(size_t)(n0 + r) * K + k0 + c];
            Pack2 pa, pw;
            pa.h[0] = (__bf16)av.x; pa.h[1] = (__bf16)av.y;
            pw.h[0] = (__bf16)wv.x; pw.h[1] = (__bf16)wv.y;
            *(unsigned*)&As[r * LDA + c] = pa.u;
            *(unsigned*)&Ws[r * LDA + c] = pw.u;
        }
        __syncthreads();
        v16bf a0 = load_a_frag_lds(&As[(wrow * 32 + l16) * LDA], khalf);
        v16bf a1 = load_a_frag_lds(&As[(wrow * 32 + 16 + l16) * LDA], khalf);
#pragma unroll
        for (int nt = 0; nt < 4; ++nt) {
            v16bf bf = load_b_frag_lds(&Ws[(wcol * 64 + nt * 16 + l16) * LDA], khalf);
            acc[0][nt] = wmma_bf16(a0, bf, acc[0][nt]);
            acc[1][nt] = wmma_bf16(a1, bf, acc[1][nt]);
        }
        __syncthreads();
    }

#pragma unroll
    for (int sub = 0; sub < 2; ++sub) {
#pragma unroll
        for (int nt = 0; nt < 4; ++nt) {
            int n = n0 + wcol * 64 + nt * 16 + l16;
            float bv = bias[n];
#pragma unroll
            for (int r = 0; r < 8; ++r) {
                int m = m0 + wrow * 32 + sub * 16 + r + khalf * 8;
                float val = acc[sub][nt][r] + bv;
                if (ACT == 1) val = fmaxf(val, 0.0f);
                if (SPLIT) {
                    int b = m / S_, s = m % S_;
                    int h = n / DK_, dk = n % DK_;
                    C[((size_t)((b * H_ + h) * S_ + s)) * DK_ + dk] = val;
                } else {
                    C[(size_t)m * N + n] = val;
                }
            }
        }
    }
}

// ---------------------------------------------------------------------------
// Attention: one workgroup per (b, h, 16-query tile).
//   scores = qk^T/sqrt(dk) (WMMA) -> dam softmax -> src-masked probs -> cumsum
//   -> distance decay -> second softmax -> P @ V (WMMA) -> concat layout out.
// ---------------------------------------------------------------------------
constexpr int SCLD = 516; // f32 stride pad: 516 % 64 == 4 -> conflict-free frag reads
constexpr int VTLD = 132;

__global__ __launch_bounds__(256) void attention_kernel(const float* __restrict__ qk,
                                                        const float* __restrict__ v,
                                                        const float* __restrict__ tb,
                                                        const float* __restrict__ gam,
                                                        float* __restrict__ out,
                                                        int bmask, int zero_pad) {
    __shared__ float sc[16 * SCLD];
    __shared__ float vt[32 * VTLD];
    const int tid = threadIdx.x, wid = tid >> 5, lane = tid & 31;
    const int khalf = lane >> 4, l16 = lane & 15;
    const int qt = blockIdx.x, h = blockIdx.y, b = blockIdx.z;
    const int q0 = qt * 16;
    const float* qbase = qk + ((size_t)(b * H_ + h) * S_) * DK_;
    const float* vbase = v  + ((size_t)(b * H_ + h) * S_) * DK_;
    const float* tbh   = tb + h * S_;
    const float scale  = 0.08838834764831844f; // 1/sqrt(128)

    // q fragments: 4 K-chunks of 32 over DK=128
    v16bf aq[4];
#pragma unroll
    for (int kc = 0; kc < 4; ++kc)
        aq[kc] = load_a_frag_f32(qbase + (size_t)(q0 + l16) * DK_ + kc * 32, khalf);

    // scores tile 16 x 512 (k == q projection since key == query everywhere)
    for (int nt = wid; nt < 32; nt += 8) {
        v8f acc = {};
#pragma unroll
        for (int kc = 0; kc < 4; ++kc) {
            v16bf bf = load_b_frag_f32(qbase + (size_t)(nt * 16 + l16) * DK_ + kc * 32, khalf);
            acc = wmma_bf16(aq[kc], bf, acc);
        }
#pragma unroll
        for (int r = 0; r < 8; ++r)
            sc[(r + khalf * 8) * SCLD + nt * 16 + l16] = acc[r] * scale;
    }
    __syncthreads();

    // per-row masked double softmax with distance decay; wave owns 2 rows
    const float gamma = -log1pf(__expf(gam[h])); // -softplus
    for (int rr = wid; rr < 16; rr += 8) {
        const int qi = q0 + rr;
        float sv[16], p[16];
        unsigned srcm = 0;
        float mx = -3.0e38f;
#pragma unroll
        for (int j = 0; j < 16; ++j) {
            int c = lane + j * 32;
            float s = sc[rr * SCLD + c];
            sv[j] = s;
            int dlt = c - qi;
            bool srcb = dlt < bmask;
            bool damb = srcb || (tbh[dlt] == 0.0f); // short-circuit guards dlt>=0
            if (srcb) srcm |= 1u << j;
            float m1 = damb ? s : NEGV;
            p[j] = m1;
            mx = fmaxf(mx, m1);
        }
        for (int off = 16; off; off >>= 1) mx = fmaxf(mx, __shfl_xor(mx, off, 32));
        float sum = 0.0f;
#pragma unroll
        for (int j = 0; j < 16; ++j) { p[j] = __expf(p[j] - mx); sum += p[j]; }
        for (int off = 16; off; off >>= 1) sum += __shfl_xor(sum, off, 32);
        float inv = 1.0f / sum;
#pragma unroll
        for (int j = 0; j < 16; ++j) p[j] = ((srcm >> j) & 1) ? p[j] * inv : 0.0f;

        // inclusive cumsum along the 512 columns (chunked wave scan)
        float carry = 0.0f;
#pragma unroll
        for (int j = 0; j < 16; ++j) {
            float x = p[j];
            for (int d = 1; d < 32; d <<= 1) {
                float t2 = __shfl_up(x, d, 32);
                if (lane >= d) x += t2;
            }
            float tot = __shfl(x, 31, 32);
            p[j] = x + carry;
            carry += tot;
        }
        const float disttot = carry;

        // distance-decayed scores + second softmax
        mx = -3.0e38f;
#pragma unroll
        for (int j = 0; j < 16; ++j) {
            int c = lane + j * 32;
            float pos = fabsf((float)(c - qi));
            float dist = sqrtf(fmaxf((disttot - p[j]) * pos, 0.0f));
            float te = __expf(dist * gamma);
            te = fminf(fmaxf(te, 1e-5f), 1e5f);
            float s2 = ((srcm >> j) & 1) ? sv[j] * te : NEGV;
            sv[j] = s2;
            mx = fmaxf(mx, s2);
        }
        for (int off = 16; off; off >>= 1) mx = fmaxf(mx, __shfl_xor(mx, off, 32));
        sum = 0.0f;
#pragma unroll
        for (int j = 0; j < 16; ++j) { sv[j] = __expf(sv[j] - mx); sum += sv[j]; }
        for (int off = 16; off; off >>= 1) sum += __shfl_xor(sum, off, 32);
        float inv2 = 1.0f / sum;
        if (zero_pad && qi == 0) inv2 = 0.0f;
#pragma unroll
        for (int j = 0; j < 16; ++j) sc[rr * SCLD + lane + j * 32] = sv[j] * inv2;
    }
    __syncthreads();

    // out(16 x 128) = P(16 x 512) @ V(512 x 128); wave owns a 16-wide dk slab
    v8f acco = {};
    for (int ck = 0; ck < 16; ++ck) {
        // stage 32x128 f32 chunk of V into LDS
#if USE_ASYNC_LDS
#pragma unroll
        for (int i = 0; i < 4; ++i) {
            int idx = tid + i * 256;        // 1024 x 16B copies
            int r = idx >> 5, c = (idx & 31) * 4;
            __builtin_amdgcn_global_load_async_to_lds_b128(
                ASYNC_GPTR(vbase + (size_t)(ck * 32 + r) * DK_ + c),
                ASYNC_LPTR(&vt[r * VTLD + c]), 0, 0);
        }
        __builtin_amdgcn_s_wait_asynccnt(0);
#else
#pragma unroll
        for (int i = 0; i < 4; ++i) {
            int idx = tid + i * 256;
            int r = idx >> 5, c = (idx & 31) * 4;
            *(float4*)&vt[r * VTLD + c] =
                *(const float4*)&vbase[(size_t)(ck * 32 + r) * DK_ + c];
        }
#endif
        __syncthreads();
        Frag fa;
        {
            const float* rp = &sc[l16 * SCLD + ck * 32];
#pragma unroll
            for (int i = 0; i < 8; ++i) {
                fa.h[i]     = (__bf16)rp[khalf * 8 + i];
                fa.h[8 + i] = (__bf16)rp[16 + khalf * 8 + i];
            }
        }
        Frag fb;
        {
            const int d = wid * 16 + l16;
#pragma unroll
            for (int i = 0; i < 8; ++i) {
                fb.h[i]     = (__bf16)vt[(khalf * 8 + i) * VTLD + d];
                fb.h[8 + i] = (__bf16)vt[(16 + khalf * 8 + i) * VTLD + d];
            }
        }
        acco = wmma_bf16(fa.v, fb.v, acco);
        __syncthreads();
    }

    // store into concat layout (B, S, D): out[b, q0+m, h*DK + d]
#pragma unroll
    for (int r = 0; r < 8; ++r) {
        int m = r + khalf * 8;
        int d = wid * 16 + l16;
        out[((size_t)(b * S_ + q0 + m)) * D_ + h * DK_ + d] = acco[r];
    }
}

// ---------------------------------------------------------------------------
// Fused residual + LayerNorm over D=1024; one row per 256-thread workgroup
// ---------------------------------------------------------------------------
__global__ __launch_bounds__(256) void ln_kernel(const float* __restrict__ x,
                                                 const float* __restrict__ r,
                                                 const float* __restrict__ g,
                                                 const float* __restrict__ bta,
                                                 float* __restrict__ out) {
    __shared__ float red[16];
    const int row = blockIdx.x, tid = threadIdx.x;
    const int wid = tid >> 5, lane = tid & 31;
    const float* xp = x + (size_t)row * D_;
    const float* rp = r + (size_t)row * D_;
    float v4[4];
    float s = 0.0f, s2 = 0.0f;
#pragma unroll
    for (int i = 0; i < 4; ++i) {
        float val = xp[tid * 4 + i] + rp[tid * 4 + i];
        v4[i] = val;
        s += val;
        s2 += val * val;
    }
    for (int off = 16; off; off >>= 1) {
        s  += __shfl_xor(s, off, 32);
        s2 += __shfl_xor(s2, off, 32);
    }
    if (lane == 0) { red[wid] = s; red[8 + wid] = s2; }
    __syncthreads();
    float ts = 0.0f, ts2 = 0.0f;
#pragma unroll
    for (int i = 0; i < 8; ++i) { ts += red[i]; ts2 += red[8 + i]; }
    const float mu   = ts / (float)D_;
    const float var  = ts2 / (float)D_ - mu * mu;
    const float rinv = rsqrtf(var + 1e-5f);
#pragma unroll
    for (int i = 0; i < 4; ++i) {
        int c = tid * 4 + i;
        out[(size_t)row * D_ + c] = (v4[i] - mu) * rinv * g[c] + bta[c];
    }
}

// ---------------------------------------------------------------------------
// Host orchestration
// ---------------------------------------------------------------------------
extern "C" void kernel_launch(void* const* d_in, const int* in_sizes, int n_in,
                              void* d_out, int out_size, void* d_ws, size_t ws_size,
                              hipStream_t stream) {
    const float* q_embed  = (const float*)d_in[0];
    const float* qa_embed = (const float*)d_in[1];
    const float* kW   = (const float*)d_in[2];
    const float* kb   = (const float*)d_in[3];
    const float* vW   = (const float*)d_in[4];
    const float* vb   = (const float*)d_in[5];
    const float* oW   = (const float*)d_in[6];
    const float* ob   = (const float*)d_in[7];
    const float* gam  = (const float*)d_in[8];
    const float* alph = (const float*)d_in[9];
    const float* ln1g = (const float*)d_in[10];
    const float* ln1b = (const float*)d_in[11];
    const float* w1   = (const float*)d_in[12];
    const float* b1   = (const float*)d_in[13];
    const float* w2   = (const float*)d_in[14];
    const float* b2   = (const float*)d_in[15];
    const float* ln2g = (const float*)d_in[16];
    const float* ln2b = (const float*)d_in[17];
    const float* gumE = (const float*)d_in[18];
    float* outp = (float*)d_out;

    // workspace layout (floats)
    float* wsf = (float*)d_ws;
    size_t o = 0;
    const size_t NBS = (size_t)B_ * S_ * D_; // 8.39M
    float* t_all = wsf + o; o += (size_t)L_ * H_ * S_;
    float* bufA  = wsf + o; o += NBS;                   // q/k proj -> o proj out -> ffn2 out
    float* bufB  = wsf + o; o += NBS;                   // v proj
    float* bufC  = wsf + o; o += NBS;                   // attention concat -> LN1 out
    float* bufF  = wsf + o; o += (size_t)MM_ * DFF_;    // ffn hidden
    float* bufY  = wsf + o; o += NBS;                   // layer0 output
    float* bufX1 = wsf + o; o += NBS;                   // layer1 output
    (void)ws_size; (void)in_sizes; (void)n_in; (void)out_size;

    const dim3 blk(256);
    const dim3 gemmDD(D_ / BN, MM_ / BM);
    const dim3 gemmFF(DFF_ / BN, MM_ / BM);
    const dim3 attG(S_ / 16, H_, B_);

    dam_kernel<<<(L_ * H_ * S_ + 255) / 256, blk, 0, stream>>>(alph, gumE, t_all,
                                                               L_ * H_ * S_);

    auto run_layer = [&](int l, int bmask, const float* query, const float* values,
                         bool apply_pos, float* out_x) {
        // q/k projection (shared weights; key == query in every call) -> head-split
        gemm_kernel<0, 1><<<gemmDD, blk, 0, stream>>>(query, kW + (size_t)l * D_ * D_,
                                                      kb + l * D_, bufA, MM_, D_, D_);
        // v projection -> head-split
        gemm_kernel<0, 1><<<gemmDD, blk, 0, stream>>>(values, vW + (size_t)l * D_ * D_,
                                                      vb + l * D_, bufB, MM_, D_, D_);
        attention_kernel<<<attG, blk, 0, stream>>>(bufA, bufB, t_all + (size_t)l * H_ * S_,
                                                   gam + l * H_, bufC, bmask,
                                                   bmask == 0 ? 1 : 0);
        // output projection
        gemm_kernel<0, 0><<<gemmDD, blk, 0, stream>>>(bufC, oW + (size_t)l * D_ * D_,
                                                      ob + l * D_, bufA, MM_, D_, D_);
        float* x1p = apply_pos ? bufC : out_x;
        ln_kernel<<<MM_, blk, 0, stream>>>(query, bufA, ln1g + l * D_, ln1b + l * D_, x1p);
        if (apply_pos) {
            gemm_kernel<1, 0><<<gemmFF, blk, 0, stream>>>(x1p, w1 + (size_t)l * DFF_ * D_,
                                                          b1 + l * DFF_, bufF, MM_, DFF_, D_);
            gemm_kernel<0, 0><<<gemmDD, blk, 0, stream>>>(bufF, w2 + (size_t)l * D_ * DFF_,
                                                          b2 + l * D_, bufA, MM_, D_, DFF_);
            ln_kernel<<<MM_, blk, 0, stream>>>(x1p, bufA, ln2g + l * D_, ln2b + l * D_, out_x);
        }
    };

    // y = layer0(bmask=1, qa, qa, qa, apply_pos=True)
    run_layer(0, 1, qa_embed, qa_embed, true, bufY);
    // x1 = layer1(bmask=1, q, q, q, apply_pos=False)
    run_layer(1, 1, q_embed, q_embed, false, bufX1);
    // out = layer2(bmask=0, query=key=x1, values=y, apply_pos=True)
    run_layer(2, 0, bufX1, bufY, true, outp);
}
